// MAE_Encoder_88227218194454
// MI455X (gfx1250) — matrix operand order repaired and checked by reading, not compiled, not run on hardware
//
#include <hip/hip_runtime.h>
#include <hip/hip_bf16.h>
#include <math.h>

typedef __bf16 bf16;
typedef __attribute__((ext_vector_type(16))) __bf16 v16bf;
typedef __attribute__((ext_vector_type(8)))  float  v8f;

static constexpr int BATCH  = 4096;
static constexpr int T      = 64;
static constexpr int PDIM   = 48;
static constexpr int D      = 192;
static constexpr int L      = 12;
static constexpr int NH     = 3;
static constexpr int DH     = 64;
static constexpr int S      = 17;     // cls + 16 kept tokens
static constexpr int REMAIN = 16;
static constexpr int NMASK  = T - REMAIN;  // 48
static constexpr int NC     = 100;

// ---------------------------------------------------------------------------
// fragment loader: 16 bf16 as two 8-element b128 loads (p[0..7] and p[16..23])
// ---------------------------------------------------------------------------
static __device__ __forceinline__ v16bf load_frag(const bf16* p) {
    union { v16bf v; uint4 u[2]; } t;
    t.u[0] = *(const uint4*)(p);
    t.u[1] = *(const uint4*)(p + 16);
    return t.v;
}

// ---------------------------------------------------------------------------
// Generic bf16 WMMA GEMM: out[M x N] = A[M x K](bf16) * Bt[N x K](bf16)^T + bias
// One wave -> 16 rows x 96 cols (6 tiles of 16x16). 8 waves / 256-thread block.
// EPI: 0 = f32 out + bias + pos[(m%T)*D + n]   (patch embedding)
//      1 = f32 out + bias                      (qkv)
//      2 = f32 out += bias + acc               (residual into x)
//      3 = bf16 out = gelu(acc + bias)         (mlp hidden)
// ---------------------------------------------------------------------------
template<int EPI, int K>
__global__ __launch_bounds__(256) void gemm_wmma(
    const bf16* __restrict__ A, const bf16* __restrict__ Bt,
    const float* __restrict__ bias,
    float* __restrict__ outF, bf16* __restrict__ outB,
    const float* __restrict__ pos, int N)
{
    constexpr int NKF    = (K >= 192) ? 6 : (K / 32);
    constexpr int NCHUNK = K / (NKF * 32);

    const int lane = threadIdx.x & 31;
    const int gw   = blockIdx.x * 8 + (threadIdx.x >> 5);
    const int m0   = gw * 16;
    const int nb   = blockIdx.y * 96;
    const int lm   = lane & 15;
    const int hi   = lane >> 4;

    v8f C[6];
#pragma unroll
    for (int t = 0; t < 6; ++t)
#pragma unroll
        for (int r = 0; r < 8; ++r) C[t][r] = 0.f;

    const bf16* arow = A  + (size_t)(m0 + lm) * K + hi * 8;   // A frag base
    const bf16* brow = Bt + (size_t)(nb + lm) * K + hi * 8;   // B frag base

    for (int ch = 0; ch < NCHUNK; ++ch) {
        v16bf a[NKF];
#pragma unroll
        for (int kf = 0; kf < NKF; ++kf)
            a[kf] = load_frag(arow + ch * (NKF * 32) + kf * 32);
#pragma unroll
        for (int nt = 0; nt < 6; ++nt) {
            const bf16* bp = brow + (size_t)nt * 16 * K + ch * (NKF * 32);
#pragma unroll
            for (int kf = 0; kf < NKF; ++kf) {
                v16bf b = load_frag(bp + kf * 32);
                C[nt] = __builtin_amdgcn_wmma_f32_16x16x32_bf16(
                    false, a[kf], false, b, (short)0, C[nt], false, false);
            }
        }
    }

#pragma unroll
    for (int nt = 0; nt < 6; ++nt) {
        int n = nb + nt * 16 + lm;
        float bs = bias[n];
#pragma unroll
        for (int r = 0; r < 8; ++r) {
            int m = m0 + hi * 8 + r;
            float v = C[nt][r] + bs;
            size_t o = (size_t)m * N + n;
            if constexpr (EPI == 0) {
                outF[o] = v + pos[(m & (T - 1)) * D + n];
            } else if constexpr (EPI == 1) {
                outF[o] = v;
            } else if constexpr (EPI == 2) {
                outF[o] += v;
            } else {
                float g = 0.5f * v * (1.0f + erff(v * 0.70710678118654752f));
                outB[o] = (bf16)g;
            }
        }
    }
}

// ---------------------------------------------------------------------------
// LayerNorm over D=192, one wave per row (wave32, 6 elems/lane).
// MODE 0: bf16 out at [row][D]. MODE 1: f32 feats out at [(s*B + b)][D].
// ---------------------------------------------------------------------------
template<int MODE>
__global__ __launch_bounds__(256) void ln_kernel(
    const float* __restrict__ X, const float* __restrict__ G,
    const float* __restrict__ Bv, bf16* __restrict__ outB,
    float* __restrict__ outF)
{
    const int lane = threadIdx.x & 31;
    const int row  = blockIdx.x * 8 + (threadIdx.x >> 5);
    const float* xr = X + (size_t)row * D;

    float v[6]; float s = 0.f;
#pragma unroll
    for (int i = 0; i < 6; ++i) { v[i] = xr[lane + i * 32]; s += v[i]; }
#pragma unroll
    for (int off = 16; off > 0; off >>= 1) s += __shfl_xor(s, off, 32);
    float mean = s * (1.f / D);
    float q = 0.f;
#pragma unroll
    for (int i = 0; i < 6; ++i) { float d = v[i] - mean; q += d * d; }
#pragma unroll
    for (int off = 16; off > 0; off >>= 1) q += __shfl_xor(q, off, 32);
    float inv = rsqrtf(q * (1.f / D) + 1e-5f);

#pragma unroll
    for (int i = 0; i < 6; ++i) {
        int d = lane + i * 32;
        float o = (v[i] - mean) * inv * G[d] + Bv[d];
        if constexpr (MODE == 0) {
            outB[(size_t)row * D + d] = (bf16)o;
        } else {
            int b = row / S, ss = row - b * S;
            outF[((size_t)ss * BATCH + b) * D + d] = o;
        }
    }
}

// ---------------------------------------------------------------------------
// Attention: one 64-thread block per (b, head). S=17, Dh=64, f32 math.
// ---------------------------------------------------------------------------
__global__ __launch_bounds__(64) void attn_kernel(
    const float* __restrict__ qkv, bf16* __restrict__ attnout)
{
    const int b = blockIdx.x / NH;
    const int h = blockIdx.x - b * NH;
    __shared__ float q[S][DH], k[S][DH], v[S][DH], sc[S][S];
    const float* base = qkv + (size_t)b * S * (3 * D);

    for (int i = threadIdx.x; i < S * DH; i += 64) {
        int s = i >> 6, d = i & 63;
        q[s][d] = base[(size_t)s * (3 * D) + h * DH + d];
        k[s][d] = base[(size_t)s * (3 * D) + D + h * DH + d];
        v[s][d] = base[(size_t)s * (3 * D) + 2 * D + h * DH + d];
    }
    __syncthreads();

    for (int p = threadIdx.x; p < S * S; p += 64) {
        int i = p / S, j = p - i * S;
        float acc = 0.f;
#pragma unroll
        for (int d = 0; d < DH; ++d) acc += q[i][d] * k[j][d];
        sc[i][j] = acc * 0.125f;  // 1/sqrt(64)
    }
    __syncthreads();

    if (threadIdx.x < S) {
        int i = threadIdx.x;
        float mx = -3.4e38f;
        for (int j = 0; j < S; ++j) mx = fmaxf(mx, sc[i][j]);
        float sum = 0.f;
        for (int j = 0; j < S; ++j) { float e = expf(sc[i][j] - mx); sc[i][j] = e; sum += e; }
        float r = 1.f / sum;
        for (int j = 0; j < S; ++j) sc[i][j] *= r;
    }
    __syncthreads();

    for (int p = threadIdx.x; p < S * DH; p += 64) {
        int i = p >> 6, d = p & 63;
        float acc = 0.f;
#pragma unroll
        for (int j = 0; j < S; ++j) acc += sc[i][j] * v[j][d];
        attnout[((size_t)b * S + i) * D + h * DH + d] = (bf16)acc;
    }
}

// ---------------------------------------------------------------------------
// Patchify: aspatches[b][t][pd] from img[b][c][h][w]
// ---------------------------------------------------------------------------
__global__ __launch_bounds__(256) void patchify_kernel(
    const float* __restrict__ img, float* __restrict__ asp)
{
    size_t idx = (size_t)blockIdx.x * 256 + threadIdx.x; // B*T*PDIM
    int pd = idx % PDIM; size_t r = idx / PDIM;
    int t = r % T; int b = r / T;
    int c = pd % 3, pp = pd / 3, py = pp >> 2, px = pp & 3;
    int hy = t >> 3, wx = t & 7;
    asp[idx] = img[(((size_t)b * 3 + c) * 32 + hy * 4 + py) * 32 + wx * 4 + px];
}

// deterministic per-class segment sums (no float atomics)
__global__ __launch_bounds__(256) void segsum_kernel(
    const float* __restrict__ asp, const int* __restrict__ label,
    float* __restrict__ sOut, float* __restrict__ sqOut)
{
    int idx = blockIdx.x * 256 + threadIdx.x;  // NC*T*PDIM = 307200
    if (idx >= NC * T * PDIM) return;
    int cls = idx / (T * PDIM);
    int tp  = idx - cls * (T * PDIM);
    float s = 0.f, sq = 0.f;
    for (int b = 0; b < BATCH; ++b) {
        if (label[b] == cls) {
            float v = asp[(size_t)b * T * PDIM + tp];
            s += v; sq += v * v;
        }
    }
    sOut[idx] = s; sqOut[idx] = sq;
}

__global__ void cnt_kernel(const int* __restrict__ label, float* __restrict__ cnt)
{
    int c = blockIdx.x * blockDim.x + threadIdx.x;
    if (c >= NC) return;
    float n = 0.f;
    for (int b = 0; b < BATCH; ++b) if (label[b] == c) n += 1.f;
    cnt[c] = n;
}

// per (class,t): pixel_var (overwrites s) and vars_c
__global__ __launch_bounds__(64) void stats_kernel(
    float* __restrict__ s, const float* __restrict__ sq,
    const float* __restrict__ cnt, float* __restrict__ varsc)
{
    int ct = blockIdx.x;           // NC*T
    int c0 = ct / T;
    float cn = cnt[c0];
    __shared__ float red[64];
    int tid = threadIdx.x;
    float var = 0.f;
    if (tid < PDIM) {
        size_t i = (size_t)ct * PDIM + tid;
        float mean = s[i] / fmaxf(cn, 1.f);
        var = (sq[i] - cn * mean * mean) / fmaxf(cn - 1.f, 1.f);
        s[i] = (cn > 1.f) ? var : 1e-6f;
    }
    red[tid] = (tid < PDIM) ? var : 0.f;
    __syncthreads();
    for (int off = 32; off > 0; off >>= 1) {
        if (tid < off) red[tid] += red[tid + off];
        __syncthreads();
    }
    if (tid == 0) varsc[ct] = (cn > 1.f) ? red[0] * (1.f / PDIM) : 1e-6f;
}

// gather stat outputs: pixel_var_patch and var1
__global__ __launch_bounds__(256) void statout_kernel(
    const float* __restrict__ pv, const float* __restrict__ varsc,
    const int* __restrict__ label, float* __restrict__ out3,
    float* __restrict__ out4)
{
    size_t idx = (size_t)blockIdx.x * 256 + threadIdx.x;  // B*T*PDIM
    int pd = idx % PDIM; size_t bt = idx / PDIM;
    int t = bt % T; int b = bt / T;
    int cls = label[b];
    out3[idx] = pv[((size_t)cls * T + t) * PDIM + pd];
    out4[idx] = varsc[(size_t)cls * T + t];
}

// gating MLPs (48x48 sigmoid -> 48x1) and PAM score, one thread per (b,t)
__global__ __launch_bounds__(256) void gate_kernel(
    const float* __restrict__ asp,
    const float* __restrict__ Wg1, const float* __restrict__ bg1,
    const float* __restrict__ Wg2, const float* __restrict__ bg2,
    const float* __restrict__ Wb1, const float* __restrict__ bb1,
    const float* __restrict__ Wb2, const float* __restrict__ bb2,
    float* __restrict__ pam)
{
    int idx = blockIdx.x * 256 + threadIdx.x;  // B*T
    const float* a = asp + (size_t)idx * PDIM;
    float av[PDIM];
#pragma unroll
    for (int i = 0; i < PDIM; ++i) av[i] = a[i];
    float g = bg2[0], bb = bb2[0];
    for (int j = 0; j < PDIM; ++j) {
        float s1 = bg1[j], s2 = bb1[j];
        for (int i = 0; i < PDIM; ++i) {
            s1 += av[i] * Wg1[i * PDIM + j];
            s2 += av[i] * Wb1[i * PDIM + j];
        }
        g  += Wg2[j] / (1.f + expf(-s1));
        bb += Wb2[j] / (1.f + expf(-s2));
    }
    float gm1 = g - 1.f, acc = 0.f;
#pragma unroll
    for (int i = 0; i < PDIM; ++i) acc += fabsf(av[i] * gm1 + bb);
    pam[idx] = acc * (1.f / PDIM);
}

// stable ascending argsort of 64 keys per sample (insertion sort)
__global__ __launch_bounds__(256) void argsort_kernel(
    const float* __restrict__ pam, int* __restrict__ order)
{
    int b = blockIdx.x * 256 + threadIdx.x;
    if (b >= BATCH) return;
    float key[T]; int idx[T];
    const float* p = pam + (size_t)b * T;
    for (int i = 0; i < T; ++i) {
        float kv = p[i];
        int j = i;
        while (j > 0 && key[j - 1] > kv) { key[j] = key[j - 1]; idx[j] = idx[j - 1]; --j; }
        key[j] = kv; idx[j] = i;
    }
    int* o = order + (size_t)b * T;
    for (int i = 0; i < T; ++i) o[i] = idx[i];
}

// token gather: build x = [cls, useful] and write mask_patches output
__global__ __launch_bounds__(192) void gather_kernel(
    const float* __restrict__ emb, const int* __restrict__ order,
    const float* __restrict__ cls, float* __restrict__ x,
    float* __restrict__ outMask)
{
    int bi = blockIdx.x;              // B * 65
    int b = bi / (S + NMASK);         // 65 slots
    int sl = bi - b * (S + NMASK);
    int d = threadIdx.x;
    if (sl == 0) {
        x[((size_t)b * S) * D + d] = cls[d];
    } else if (sl <= REMAIN) {
        int t = order[(size_t)b * T + (sl - 1)];
        x[((size_t)b * S + sl) * D + d] = emb[((size_t)b * T + t) * D + d];
    } else {
        int j = sl - (REMAIN + 1);
        int t = order[(size_t)b * T + REMAIN + j];
        outMask[((size_t)j * BATCH + b) * D + d] = emb[((size_t)b * T + t) * D + d];
    }
}

// aspatches f32 -> bf16 padded to K=64
__global__ __launch_bounds__(256) void aspad_kernel(
    const float* __restrict__ asp, bf16* __restrict__ asbf)
{
    size_t idx = (size_t)blockIdx.x * 256 + threadIdx.x;  // B*T*64
    int k = idx & 63; size_t m = idx >> 6;
    asbf[idx] = (k < PDIM) ? (bf16)asp[m * PDIM + k] : (bf16)0.f;
}

// weight transpose + f32->bf16:  src[K x N] -> dst[N x Kp] (zero-pad K..Kp)
__global__ __launch_bounds__(256) void wtrans_kernel(
    const float* __restrict__ src, bf16* __restrict__ dst, int K, int N, int Kp)
{
    int idx = blockIdx.x * 256 + threadIdx.x;
    if (idx >= N * Kp) return;
    int k = idx % Kp; int n = idx / Kp;
    dst[idx] = (k < K) ? (bf16)src[(size_t)k * N + n] : (bf16)0.f;
}

// ---------------------------------------------------------------------------
extern "C" void kernel_launch(void* const* d_in, const int* in_sizes, int n_in,
                              void* d_out, int out_size, void* d_ws, size_t ws_size,
                              hipStream_t stream)
{
    (void)in_sizes; (void)n_in; (void)out_size; (void)ws_size;
    const float* img  = (const float*)d_in[0];
    const float* cls  = (const float*)d_in[1];
    const float* pos  = (const float*)d_in[2];
    const float* Wp   = (const float*)d_in[3];
    const float* bp   = (const float*)d_in[4];
    const float* Wg1  = (const float*)d_in[5];
    const float* bg1  = (const float*)d_in[6];
    const float* Wg2  = (const float*)d_in[7];
    const float* bg2  = (const float*)d_in[8];
    const float* Wb1  = (const float*)d_in[9];
    const float* bb1  = (const float*)d_in[10];
    const float* Wb2  = (const float*)d_in[11];
    const float* bb2  = (const float*)d_in[12];
    const float* ln1g = (const float*)d_in[13];
    const float* ln1b = (const float*)d_in[14];
    const float* Wqkv = (const float*)d_in[15];
    const float* bqkv = (const float*)d_in[16];
    const float* Wo   = (const float*)d_in[17];
    const float* bo   = (const float*)d_in[18];
    const float* ln2g = (const float*)d_in[19];
    const float* ln2b = (const float*)d_in[20];
    const float* Wm1  = (const float*)d_in[21];
    const float* bm1  = (const float*)d_in[22];
    const float* Wm2  = (const float*)d_in[23];
    const float* bm2  = (const float*)d_in[24];
    const float* lnfg = (const float*)d_in[25];
    const float* lnfb = (const float*)d_in[26];
    const int*   label= (const int*)d_in[27];

    // workspace layout
    char* w = (char*)d_ws;
    size_t off = 0;
    auto alloc = [&](size_t bytes) -> void* {
        void* p = w + off; off += (bytes + 255) & ~(size_t)255; return p;
    };
    float* aspatch = (float*)alloc((size_t)BATCH * T * PDIM * 4);
    float* segS    = (float*)alloc((size_t)NC * T * PDIM * 4);
    float* segSQ   = (float*)alloc((size_t)NC * T * PDIM * 4);
    float* cnt     = (float*)alloc((size_t)NC * 4);
    float* varsc   = (float*)alloc((size_t)NC * T * 4);
    float* pam     = (float*)alloc((size_t)BATCH * T * 4);
    int*   order   = (int*)  alloc((size_t)BATCH * T * 4);
    bf16*  asbf    = (bf16*) alloc((size_t)BATCH * T * 64 * 2);
    constexpr size_t PERL = (size_t)576*192 + 192*192 + 768*192 + 192*768;
    bf16*  wpT     = (bf16*) alloc((192 * 64 + PERL * L) * 2);
    bf16*  lbase   = wpT + 192 * 64;
    float* x       = (float*)alloc((size_t)BATCH * S * D * 4);
    bf16*  hbf     = (bf16*) alloc((size_t)BATCH * S * D * 2);
    bf16*  attnbf  = (bf16*) alloc((size_t)BATCH * S * D * 2);
    void*  big     = alloc((size_t)BATCH * T * D * 4);   // emb / qkv / mlph reuse
    float* emb  = (float*)big;
    float* qkv  = (float*)big;
    bf16*  mlph = (bf16*)big;

    // outputs
    float* outF     = (float*)d_out;
    float* outFeats = outF;
    float* outMask  = outFeats + (size_t)S * BATCH * D;
    float* outPV    = outMask + (size_t)NMASK * BATCH * D;
    float* outV1    = outPV + (size_t)BATCH * T * PDIM;

    // ---- weight conversion (bf16 transposed) ----
    auto wt = [&](const float* src, bf16* dst, int K, int N, int Kp) {
        int n = N * Kp;
        wtrans_kernel<<<(n + 255) / 256, 256, 0, stream>>>(src, dst, K, N, Kp);
    };
    wt(Wp, wpT, PDIM, D, 64);
    for (int l = 0; l < L; ++l) {
        bf16* qkvT = lbase + l * PERL;
        bf16* woT  = qkvT + 576 * 192;
        bf16* m1T  = woT + 192 * 192;
        bf16* m2T  = m1T + 768 * 192;
        wt(Wqkv + (size_t)l * 192 * 576, qkvT, 192, 576, 192);
        wt(Wo   + (size_t)l * 192 * 192, woT,  192, 192, 192);
        wt(Wm1  + (size_t)l * 192 * 768, m1T,  192, 768, 192);
        wt(Wm2  + (size_t)l * 768 * 192, m2T,  768, 192, 768);
    }

    // ---- patchify + statistics ----
    patchify_kernel<<<(BATCH * T * PDIM) / 256, 256, 0, stream>>>(img, aspatch);
    segsum_kernel<<<(NC * T * PDIM + 255) / 256, 256, 0, stream>>>(aspatch, label, segS, segSQ);
    cnt_kernel<<<1, 128, 0, stream>>>(label, cnt);
    stats_kernel<<<NC * T, 64, 0, stream>>>(segS, segSQ, cnt, varsc);
    statout_kernel<<<(BATCH * T * PDIM) / 256, 256, 0, stream>>>(segS, varsc, label, outPV, outV1);

    // ---- gating, argsort ----
    gate_kernel<<<(BATCH * T) / 256, 256, 0, stream>>>(
        aspatch, Wg1, bg1, Wg2, bg2, Wb1, bb1, Wb2, bb2, pam);
    argsort_kernel<<<BATCH / 256, 256, 0, stream>>>(pam, order);

    // ---- patch embedding (WMMA) + pos, then gather tokens ----
    aspad_kernel<<<(BATCH * T * 64) / 256, 256, 0, stream>>>(aspatch, asbf);
    gemm_wmma<0, 64><<<dim3((BATCH * T) / 128, D / 96), 256, 0, stream>>>(
        asbf, wpT, bp, emb, nullptr, pos, D);
    gather_kernel<<<BATCH * (S + NMASK), 192, 0, stream>>>(emb, order, cls, x, outMask);

    // ---- transformer ----
    const int MR = BATCH * S;            // 69632 rows
    const dim3 gLN(MR / 8);
    for (int l = 0; l < L; ++l) {
        bf16* qkvT = lbase + l * PERL;
        bf16* woT  = qkvT + 576 * 192;
        bf16* m1T  = woT + 192 * 192;
        bf16* m2T  = m1T + 768 * 192;

        ln_kernel<0><<<gLN, 256, 0, stream>>>(x, ln1g + l * D, ln1b + l * D, hbf, nullptr);
        gemm_wmma<1, 192><<<dim3(MR / 128, 576 / 96), 256, 0, stream>>>(
            hbf, qkvT, bqkv + l * 576, qkv, nullptr, nullptr, 576);
        attn_kernel<<<BATCH * NH, 64, 0, stream>>>(qkv, attnbf);
        gemm_wmma<2, 192><<<dim3(MR / 128, D / 96), 256, 0, stream>>>(
            attnbf, woT, bo + l * D, x, nullptr, nullptr, D);

        ln_kernel<0><<<gLN, 256, 0, stream>>>(x, ln2g + l * D, ln2b + l * D, hbf, nullptr);
        gemm_wmma<3, 192><<<dim3(MR / 128, 768 / 96), 256, 0, stream>>>(
            hbf, m1T, bm1 + l * 768, nullptr, mlph, nullptr, 768);
        gemm_wmma<2, 768><<<dim3(MR / 128, D / 96), 256, 0, stream>>>(
            mlph, m2T, bm2 + l * D, x, nullptr, nullptr, D);
    }

    // ---- final LN -> feats (transposed to (S, B, D)) ----
    ln_kernel<1><<<gLN, 256, 0, stream>>>(x, lnfg, lnfb, nullptr, outFeats);
}
